// Attention_35527969473061
// MI455X (gfx1250) — matrix operand order repaired and checked by reading, not compiled, not run on hardware
//
#include <hip/hip_runtime.h>
#include <hip/hip_bf16.h>
#include <math.h>

// ---------------------------------------------------------------------------
// Causal self-attention, L=8192, D=D_ATTN=D_V=2048, fp32 in/out.
// All matmuls on v_wmma_f32_16x16x32_bf16 (wave32, CDNA5).
// ---------------------------------------------------------------------------

typedef __attribute__((ext_vector_type(16))) __bf16 v16bf;
typedef __attribute__((ext_vector_type(8)))  __bf16 bf16x8;
typedef __attribute__((ext_vector_type(8)))  float  v8f;

#define LSEQ   8192
#define DMODEL 2048
static constexpr float kScale = 0.022097086912079608f; // 1/sqrt(2048)

// ---- WMMA fragment loaders (layouts per CDNA5 ISA 7.12.2) ------------------

// 16-bit A matrix 16x32 (MxK): lane m = lane&15, kb = (lane>>4)*8,
// holds K = [kb, kb+8) and [kb+16, kb+24)  -> two contiguous 16B runs.
__device__ __forceinline__ v16bf load_a_bf16(const __bf16* base, int ld,
                                             int m0, int k0, int lane) {
  int m  = m0 + (lane & 15);
  int kb = (lane >> 4) * 8;
  const __bf16* p = base + (size_t)m * ld + k0 + kb;
  bf16x8 lo = *(const bf16x8*)(p);
  bf16x8 hi = *(const bf16x8*)(p + 16);
  return __builtin_shufflevector(lo, hi, 0,1,2,3,4,5,6,7,8,9,10,11,12,13,14,15);
}

// 16-bit B matrix 32x16 (KxN): lane n = lane&15, kb = (lane>>4)*8,
// holds K = [kb, kb+8) and [kb+16, kb+24) of column n (source K-major).
__device__ __forceinline__ v16bf load_b_bf16(const __bf16* base, int ld,
                                             int n0, int k0, int lane) {
  int n  = n0 + (lane & 15);
  int kb = (lane >> 4) * 8;
  const __bf16* p = base + (size_t)n * ld + k0 + kb;
  bf16x8 lo = *(const bf16x8*)(p);
  bf16x8 hi = *(const bf16x8*)(p + 16);
  return __builtin_shufflevector(lo, hi, 0,1,2,3,4,5,6,7,8,9,10,11,12,13,14,15);
}

// A fragment straight from an f32 row-major matrix (converted on the fly).
__device__ __forceinline__ v16bf load_a_f32(const float* base, int ld,
                                            int m0, int k0, int lane) {
  int m  = m0 + (lane & 15);
  int kb = (lane >> 4) * 8;
  const float* p = base + (size_t)m * ld + k0 + kb;
  float4 f0 = *(const float4*)(p);
  float4 f1 = *(const float4*)(p + 4);
  float4 f2 = *(const float4*)(p + 16);
  float4 f3 = *(const float4*)(p + 20);
  v16bf a;
  a[0]=(__bf16)f0.x;  a[1]=(__bf16)f0.y;  a[2]=(__bf16)f0.z;  a[3]=(__bf16)f0.w;
  a[4]=(__bf16)f1.x;  a[5]=(__bf16)f1.y;  a[6]=(__bf16)f1.z;  a[7]=(__bf16)f1.w;
  a[8]=(__bf16)f2.x;  a[9]=(__bf16)f2.y;  a[10]=(__bf16)f2.z; a[11]=(__bf16)f2.w;
  a[12]=(__bf16)f3.x; a[13]=(__bf16)f3.y; a[14]=(__bf16)f3.z; a[15]=(__bf16)f3.w;
  return a;
}

__device__ __forceinline__ v8f wmma_bf16(v16bf a, v16bf b, v8f c) {
  // D = A*B + C, 16x16x32 bf16 -> f32
  return __builtin_amdgcn_wmma_f32_16x16x32_bf16(false, a, false, b,
                                                 (short)0, c, false, false);
}

// ---------------------------------------------------------------------------
// Kernel 0: W (D x Dout, f32, row-major) -> W^T (Dout x D, bf16, row-major)
// ---------------------------------------------------------------------------
__global__ void __launch_bounds__(256)
wt_transpose_cvt(const float* __restrict__ in, __bf16* __restrict__ out) {
  __shared__ float tile[32][33];
  int x  = blockIdx.x * 32 + threadIdx.x;
  int y0 = blockIdx.y * 32 + threadIdx.y;
#pragma unroll
  for (int j = 0; j < 32; j += 8)
    tile[threadIdx.y + j][threadIdx.x] = in[(size_t)(y0 + j) * DMODEL + x];
  __syncthreads();
  int xo = blockIdx.y * 32 + threadIdx.x;
  int yo = blockIdx.x * 32 + threadIdx.y;
#pragma unroll
  for (int j = 0; j < 32; j += 8)
    out[(size_t)(yo + j) * DMODEL + xo] = (__bf16)tile[threadIdx.x][threadIdx.y + j];
}

// ---------------------------------------------------------------------------
// Kernel 1: projection GEMM  out = X (L x D, f32) @ W (D x D) + b  -> bf16
// Each wave: one 16x64 output tile (4 accumulators), k-loop over D in 32s.
// TRANS_OUT=true writes out[col*L + row] (for V^T), else row-major.
// ---------------------------------------------------------------------------
template <bool TRANS_OUT>
__global__ void __launch_bounds__(256)
proj_gemm(const float* __restrict__ X, const __bf16* __restrict__ WT,
          const float* __restrict__ bias, __bf16* __restrict__ out) {
  const int lane = threadIdx.x & 31;
  const int wv   = threadIdx.x >> 5;
  const int gid  = blockIdx.x * 8 + wv;   // 16384 wave-tiles total
  const int mt   = gid >> 5;              // 512 m-tiles of 16
  const int nt   = gid & 31;              // 32 n-tiles of 64
  const int m0 = mt * 16, n0 = nt * 64;

  v8f acc[4];
#pragma unroll
  for (int t = 0; t < 4; ++t)
#pragma unroll
    for (int i = 0; i < 8; ++i) acc[t][i] = 0.0f;

  for (int k0 = 0; k0 < DMODEL; k0 += 32) {
    v16bf a = load_a_f32(X, DMODEL, m0, k0, lane);
#pragma unroll
    for (int t = 0; t < 4; ++t) {
      v16bf b = load_b_bf16(WT, DMODEL, n0 + t * 16, k0, lane);
      acc[t] = wmma_bf16(a, b, acc[t]);
    }
  }

  const int half = lane >> 4, nl = lane & 15;
#pragma unroll
  for (int t = 0; t < 4; ++t) {
    int col = n0 + t * 16 + nl;
    float bv = bias[col];
#pragma unroll
    for (int r = 0; r < 8; ++r) {
      int row = m0 + r + 8 * half;
      float v = acc[t][r] + bv;
      if (TRANS_OUT) out[(size_t)col * LSEQ + row]   = (__bf16)v;
      else           out[(size_t)row * DMODEL + col] = (__bf16)v;
    }
  }
}

// ---------------------------------------------------------------------------
// Kernel 2: softmax stats.  One wave per 16-row q-tile; stream kv in 64-wide
// tiles (4 C fragments) so each Q (A) fragment is amortized over 4 WMMAs.
// C-layout: lane holds column (c + lane&15) for 8 rows (r + 8*(lane>>4)).
// Per-lane online (m,l) over its columns, 16-lane LSE merge at the end.
// ---------------------------------------------------------------------------
__global__ void __launch_bounds__(256)
attn_stats(const __bf16* __restrict__ Qb, const __bf16* __restrict__ Kb,
           float* __restrict__ Mrow, float* __restrict__ Lrow) {
  const int lane = threadIdx.x & 31;
  const int wv   = threadIdx.x >> 5;
  const int qt   = blockIdx.x * 8 + wv;   // 512 q-tiles
  const int m0   = qt * 16;
  const int half = lane >> 4, nl = lane & 15;

  float mr[8], lr[8];
#pragma unroll
  for (int r = 0; r < 8; ++r) { mr[r] = -INFINITY; lr[r] = 0.0f; }

  for (int kv0 = 0; kv0 < m0 + 16; kv0 += 64) {
    v8f s[4];
#pragma unroll
    for (int t = 0; t < 4; ++t)
#pragma unroll
      for (int i = 0; i < 8; ++i) s[t][i] = 0.0f;

#pragma unroll 2
    for (int k0 = 0; k0 < DMODEL; k0 += 32) {
      v16bf a = load_a_bf16(Qb, DMODEL, m0, k0, lane);   // Q rows (shared)
#pragma unroll
      for (int t = 0; t < 4; ++t) {
        v16bf b = load_b_bf16(Kb, DMODEL, kv0 + t * 16, k0, lane);
        s[t] = wmma_bf16(a, b, s[t]);
      }
    }

#pragma unroll
    for (int t = 0; t < 4; ++t) {
      int col = kv0 + t * 16 + nl;
#pragma unroll
      for (int r = 0; r < 8; ++r) {
        int row = m0 + r + 8 * half;
        if (col <= row) {                      // causal mask
          float sv = s[t][r] * kScale;
          float mn = fmaxf(mr[r], sv);
          float e_old = (mr[r] == -INFINITY) ? 0.0f : __expf(mr[r] - mn);
          lr[r] = lr[r] * e_old + __expf(sv - mn);
          mr[r] = mn;
        }
      }
    }
  }

  // merge 16 column-lanes within each half-wave (guarded LSE combine)
#pragma unroll
  for (int r = 0; r < 8; ++r) {
    float m = mr[r], l = lr[r];
#pragma unroll
    for (int off = 8; off >= 1; off >>= 1) {
      float mo = __shfl_xor(m, off, 16);
      float lo = __shfl_xor(l, off, 16);
      float mn = fmaxf(m, mo);
      float e0 = (m  == -INFINITY) ? 0.0f : __expf(m  - mn);
      float e1 = (mo == -INFINITY) ? 0.0f : __expf(mo - mn);
      l = l * e0 + lo * e1;
      m = mn;
    }
    if (nl == 0) {
      int row = m0 + r + 8 * half;
      Mrow[row] = m;
      Lrow[row] = l;
    }
  }
}

// ---------------------------------------------------------------------------
// Kernel 3: output.  One block per 16-row q-tile, 8 waves.
// Per 256-wide kv tile: wave w computes the S chunk cols [kv0+32w, +32)
// (2 fragments, Q fragment amortized), applies exp(s*scale - M)/L, stages
// P (16x256 bf16, 8 KB) in LDS; then every wave accumulates its 256-feature
// output slice: O_w += P @ V^T-chunk (8 preloaded A frags, 128 WMMAs/tile).
// L % 256 == 0 -> no OOB loads; causality handled purely by the exp mask.
// ---------------------------------------------------------------------------
__global__ void __launch_bounds__(256)
attn_out(const __bf16* __restrict__ Qb, const __bf16* __restrict__ Kb,
         const __bf16* __restrict__ Vt, const float* __restrict__ Mrow,
         const float* __restrict__ Lrow, float* __restrict__ out) {
  __shared__ __bf16 Pt[16 * 256];
  const int lane = threadIdx.x & 31;
  const int wv   = threadIdx.x >> 5;
  const int m0   = blockIdx.x * 16;
  const int half = lane >> 4, nl = lane & 15;
  const int nf0  = wv * 256;               // this wave's feature slice

  float mrow[8], rl[8];
#pragma unroll
  for (int r = 0; r < 8; ++r) {
    int row = m0 + r + 8 * half;
    mrow[r] = Mrow[row];
    rl[r]   = 1.0f / Lrow[row];            // every row has >=1 unmasked col
  }

  v8f acc[16];
#pragma unroll
  for (int t = 0; t < 16; ++t)
#pragma unroll
    for (int i = 0; i < 8; ++i) acc[t][i] = 0.0f;

  for (int kv0 = 0; kv0 < m0 + 16; kv0 += 256) {
    // ---- S chunk for this wave: columns [c0, c0+32) ----
    const int c0 = kv0 + wv * 32;
    v8f s[2];
#pragma unroll
    for (int u = 0; u < 2; ++u)
#pragma unroll
      for (int i = 0; i < 8; ++i) s[u][i] = 0.0f;

#pragma unroll 2
    for (int k0 = 0; k0 < DMODEL; k0 += 32) {
      v16bf a = load_a_bf16(Qb, DMODEL, m0, k0, lane);
#pragma unroll
      for (int u = 0; u < 2; ++u) {
        v16bf b = load_b_bf16(Kb, DMODEL, c0 + u * 16, k0, lane);
        s[u] = wmma_bf16(a, b, s[u]);
      }
    }

    // ---- normalized probabilities into LDS (causal mask here) ----
#pragma unroll
    for (int u = 0; u < 2; ++u) {
      int col = c0 + u * 16 + nl;
#pragma unroll
      for (int r = 0; r < 8; ++r) {
        int row = m0 + r + 8 * half;
        float p = 0.0f;
        if (col <= row) p = __expf(s[u][r] * kScale - mrow[r]) * rl[r];
        Pt[(r + 8 * half) * 256 + wv * 32 + u * 16 + nl] = (__bf16)p;
      }
    }
    __syncthreads();

    // ---- O_w += P(16x256) @ V(256 x 256-slice), V^T is K-major ----
    v16bf pa[8];
#pragma unroll
    for (int j = 0; j < 8; ++j)
      pa[j] = load_a_bf16(Pt, 256, 0, j * 32, lane);   // ds_load_b128 pairs
#pragma unroll
    for (int j = 0; j < 8; ++j) {
      const int kvbase = kv0 + j * 32;
#pragma unroll
      for (int t = 0; t < 16; ++t) {
        v16bf b = load_b_bf16(Vt, LSEQ, nf0 + t * 16, kvbase, lane);
        acc[t] = wmma_bf16(pa[j], b, acc[t]);
      }
    }
    __syncthreads();  // before next iteration overwrites Pt
  }

  // ---- store f32 output ----
#pragma unroll
  for (int t = 0; t < 16; ++t) {
    int col = nf0 + t * 16 + nl;
#pragma unroll
    for (int r = 0; r < 8; ++r) {
      int row = m0 + r + 8 * half;
      out[(size_t)row * DMODEL + col] = acc[t][r];
    }
  }
}

// ---------------------------------------------------------------------------
extern "C" void kernel_launch(void* const* d_in, const int* in_sizes, int n_in,
                              void* d_out, int out_size, void* d_ws, size_t ws_size,
                              hipStream_t stream) {
  const float* x  = (const float*)d_in[0];
  const float* z  = (const float*)d_in[1];
  const float* Wq = (const float*)d_in[2];
  const float* bq = (const float*)d_in[3];
  const float* Wk = (const float*)d_in[4];
  const float* bk = (const float*)d_in[5];
  const float* Wv = (const float*)d_in[6];
  const float* bv = (const float*)d_in[7];
  float* out = (float*)d_out;

  char* ws = (char*)d_ws;
  size_t off = 0;
  auto carve = [&](size_t bytes) -> char* {
    char* p = ws + off;
    off = (off + bytes + 255) & ~(size_t)255;
    return p;
  };
  const size_t wbytes = (size_t)DMODEL * DMODEL * sizeof(__bf16); //  8 MB
  const size_t qbytes = (size_t)LSEQ   * DMODEL * sizeof(__bf16); // 32 MB
  __bf16* WqT = (__bf16*)carve(wbytes);
  __bf16* WkT = (__bf16*)carve(wbytes);
  __bf16* WvT = (__bf16*)carve(wbytes);
  __bf16* Qb  = (__bf16*)carve(qbytes);         // L x D row-major
  __bf16* Kb  = (__bf16*)carve(qbytes);         // L x D row-major
  __bf16* Vt  = (__bf16*)carve(qbytes);         // D x L row-major (V^T)
  float*  Mb  = (float*)carve((size_t)LSEQ * sizeof(float));
  float*  Lb  = (float*)carve((size_t)LSEQ * sizeof(float));
  (void)ws_size; (void)in_sizes; (void)n_in; (void)out_size;

  dim3 tb(32, 8), tg(DMODEL / 32, DMODEL / 32);
  wt_transpose_cvt<<<tg, tb, 0, stream>>>(Wq, WqT);
  wt_transpose_cvt<<<tg, tb, 0, stream>>>(Wk, WkT);
  wt_transpose_cvt<<<tg, tb, 0, stream>>>(Wv, WvT);

  // 512 m-tiles * 32 n-tiles = 16384 wave-tiles / 8 waves per block
  proj_gemm<false><<<2048, 256, 0, stream>>>(x, WqT, bq, Qb);
  proj_gemm<false><<<2048, 256, 0, stream>>>(z, WkT, bk, Kb);
  proj_gemm<true ><<<2048, 256, 0, stream>>>(z, WvT, bv, Vt);

  attn_stats<<<LSEQ / 16 / 8, 256, 0, stream>>>(Qb, Kb, Mb, Lb);
  attn_out  <<<LSEQ / 16,     256, 0, stream>>>(Qb, Kb, Vt, Mb, Lb, out);
}
